// DecoderLayer_80882824118321
// MI455X (gfx1250) — matrix operand order repaired and checked by reading, not compiled, not run on hardware
//
#include <hip/hip_runtime.h>
#include <hip/hip_bf16.h>

typedef unsigned short u16;
typedef __attribute__((ext_vector_type(16))) __bf16 v16bf;
typedef __attribute__((ext_vector_type(8)))  __bf16 v8bf;
typedef __attribute__((ext_vector_type(8)))  float  v8f;
typedef __attribute__((ext_vector_type(4)))  int    i4v;

#define E_  2048
#define H_  32
#define KH_ 8
#define HD_ 64
#define KV_ 512
#define F_  8192
#define B_  2
#define S_  2048
#define M_  (B_ * S_)        // 4096 tokens
#define NQKV_ (E_ + 2 * KV_) // 3072

#if defined(__has_builtin)
#if __has_builtin(__builtin_amdgcn_global_load_async_to_lds_b128)
#define USE_ASYNC 1
#endif
#if __has_builtin(__builtin_amdgcn_update_dpp)
#define USE_DPP 1
#endif
#endif
#ifndef USE_ASYNC
#define USE_ASYNC 0
#endif
#ifndef USE_DPP
#define USE_DPP 0
#endif

static __device__ __forceinline__ u16 f2bf(float x) {
    unsigned int u = __float_as_uint(x);
    u += 0x7fffu + ((u >> 16) & 1u);
    return (u16)(u >> 16);
}

static __device__ __forceinline__ v8bf ld8(const u16* p) {
    return *reinterpret_cast<const v8bf*>(p);
}

static __device__ __forceinline__ v16bf mk16(v8bf lo, v8bf hi) {
    union { v16bf v; v8bf h[2]; } u;
    u.h[0] = lo; u.h[1] = hi;
    return u.v;
}

static __device__ __forceinline__ v8f wmma_bf(v16bf a, v16bf b, v8f c) {
    return __builtin_amdgcn_wmma_f32_16x16x32_bf16(false, a, false, b, (short)0, c, false, false);
}

#if USE_DPP
// DPP16 ROW_XMASK cross-lane move: lane reads lane^MASK within its 16-lane row (pure VALU)
template <int MASK>
static __device__ __forceinline__ float dpp_xor(float x) {
    int i = __float_as_int(x);
    return __int_as_float(__builtin_amdgcn_update_dpp(i, i, 0x160 | MASK, 0xf, 0xf, true));
}
#endif

static __device__ __forceinline__ float red16_max(float x) {
#if USE_DPP
    x = fmaxf(x, dpp_xor<1>(x));
    x = fmaxf(x, dpp_xor<2>(x));
    x = fmaxf(x, dpp_xor<4>(x));
    x = fmaxf(x, dpp_xor<8>(x));
#else
    #pragma unroll
    for (int off = 1; off < 16; off <<= 1) x = fmaxf(x, __shfl_xor(x, off, 16));
#endif
    return x;
}

static __device__ __forceinline__ float red16_sum(float x) {
#if USE_DPP
    x += dpp_xor<1>(x);
    x += dpp_xor<2>(x);
    x += dpp_xor<4>(x);
    x += dpp_xor<8>(x);
#else
    #pragma unroll
    for (int off = 1; off < 16; off <<= 1) x += __shfl_xor(x, off, 16);
#endif
    return x;
}

#if USE_ASYNC
static __device__ __forceinline__ void async_cp16(const u16* g, u16* l) {
    // global -> LDS async b128 copy (tracked by ASYNCcnt)
    __builtin_amdgcn_global_load_async_to_lds_b128((i4v*)g, (i4v*)l, 0, 0);
}
static __device__ __forceinline__ void wait_async0() {
#if __has_builtin(__builtin_amdgcn_s_wait_asynccnt)
    __builtin_amdgcn_s_wait_asynccnt(0);
#else
    asm volatile("s_wait_asynccnt 0" ::: "memory");
#endif
}
#endif

// ---------------- fp32 -> bf16 conversion (weights) ----------------
__global__ void cvt_f32_bf16(const float* __restrict__ in, u16* __restrict__ out, int n4) {
    int i = blockIdx.x * blockDim.x + threadIdx.x;
    for (; i < n4; i += gridDim.x * blockDim.x) {
        float4 v = reinterpret_cast<const float4*>(in)[i];
        u16 o[4] = { f2bf(v.x), f2bf(v.y), f2bf(v.z), f2bf(v.w) };
        reinterpret_cast<uint2*>(out)[i] = *reinterpret_cast<uint2*>(o);
    }
}

// ---------------- RMSNorm: fp32 in -> bf16 out ----------------
__global__ void rmsnorm_bf16(const float* __restrict__ x, const float* __restrict__ w,
                             u16* __restrict__ out) {
    __shared__ float red[256];
    const float* xr = x + (size_t)blockIdx.x * E_;
    int tid = threadIdx.x;
    float ss = 0.f;
    #pragma unroll
    for (int i = tid; i < E_; i += 256) { float v = xr[i]; ss += v * v; }
    red[tid] = ss; __syncthreads();
    #pragma unroll
    for (int o = 128; o > 0; o >>= 1) {
        if (tid < o) red[tid] += red[tid + o];
        __syncthreads();
    }
    float inv = rsqrtf(red[0] * (1.0f / E_) + 1e-5f);
    u16* orow = out + (size_t)blockIdx.x * E_;
    #pragma unroll
    for (int i = tid; i < E_; i += 256) orow[i] = f2bf(xr[i] * inv * w[i]);
}

// ---------------- GEMM: C[M,N] = A[M,K] @ W[N,K]^T  (bf16 in, f32 accum) ----
// MODE 0: outF = acc   MODE 1: outF = acc + res   MODE 2 (DUAL): outBf = silu(acc)*acc2
template <int MODE, bool DUAL>
__global__ __launch_bounds__(256) void gemm_bf16(
    const u16* __restrict__ A, const u16* __restrict__ Bw, const u16* __restrict__ Bw2,
    const float* __restrict__ res, float* __restrict__ outF, u16* __restrict__ outBf,
    int M, int N, int K) {
    __shared__ __align__(16) u16 As[128][72];                 // 64 data + 8 pad
    __shared__ __align__(16) u16 Bs[128][72];
    __shared__ __align__(16) u16 Bs2[DUAL ? 128 : 1][72];

    const int m0 = blockIdx.y * 128, n0 = blockIdx.x * 128;
    const int tid = threadIdx.x;
    const int wave = tid >> 5, lane = tid & 31;
    const int wm = wave >> 2, wn = wave & 3;          // 2 (M) x 4 (N) waves
    const int g = lane >> 4, lr = lane & 15;
    const int rowL = tid >> 1, kc2 = (tid & 1) * 32;  // loader: 32 bf16 per thread per tile

    v8f acc[4][2] = {};
    v8f acc2[4][2] = {};

    for (int k0 = 0; k0 < K; k0 += 64) {
        const u16* gA = &A[(size_t)(m0 + rowL) * K + k0 + kc2];
        const u16* gB = &Bw[(size_t)(n0 + rowL) * K + k0 + kc2];
        const u16* gC = DUAL ? &Bw2[(size_t)(n0 + rowL) * K + k0 + kc2] : nullptr;
#if USE_ASYNC
        #pragma unroll
        for (int q = 0; q < 4; ++q) {
            async_cp16(gA + q * 8, &As[rowL][kc2 + q * 8]);
            async_cp16(gB + q * 8, &Bs[rowL][kc2 + q * 8]);
            if (DUAL) async_cp16(gC + q * 8, &Bs2[rowL][kc2 + q * 8]);
        }
        wait_async0();
#else
        #pragma unroll
        for (int q = 0; q < 4; ++q) {
            uint4 a = reinterpret_cast<const uint4*>(gA)[q];
            *reinterpret_cast<uint4*>(&As[rowL][kc2 + q * 8]) = a;
        }
        #pragma unroll
        for (int q = 0; q < 4; ++q) {
            uint4 b = reinterpret_cast<const uint4*>(gB)[q];
            *reinterpret_cast<uint4*>(&Bs[rowL][kc2 + q * 8]) = b;
        }
        if (DUAL) {
            #pragma unroll
            for (int q = 0; q < 4; ++q) {
                uint4 c = reinterpret_cast<const uint4*>(gC)[q];
                *reinterpret_cast<uint4*>(&Bs2[rowL][kc2 + q * 8]) = c;
            }
        }
        if (k0 + 64 < K) {
            __builtin_prefetch(gA + 64, 0, 0);
            __builtin_prefetch(gB + 64, 0, 0);
        }
#endif
        __syncthreads();

        #pragma unroll
        for (int kk = 0; kk < 64; kk += 32) {
            v16bf af[4];
            #pragma unroll
            for (int s = 0; s < 4; ++s) {
                int row = wm * 64 + s * 16 + lr;
                af[s] = mk16(ld8(&As[row][kk + 16 * g]), ld8(&As[row][kk + 16 * g + 8]));
            }
            #pragma unroll
            for (int t = 0; t < 2; ++t) {
                int col = wn * 32 + t * 16 + lr;
                v16bf bfr = mk16(ld8(&Bs[col][kk + 16 * g]), ld8(&Bs[col][kk + 16 * g + 8]));
                #pragma unroll
                for (int s = 0; s < 4; ++s) acc[s][t] = wmma_bf(af[s], bfr, acc[s][t]);
                if (DUAL) {
                    v16bf bfr2 = mk16(ld8(&Bs2[col][kk + 16 * g]), ld8(&Bs2[col][kk + 16 * g + 8]));
                    #pragma unroll
                    for (int s = 0; s < 4; ++s) acc2[s][t] = wmma_bf(af[s], bfr2, acc2[s][t]);
                }
            }
        }
        __syncthreads();
    }

    // epilogue: D layout — row m = v + 8*(lane/16), col n = lane%16
    #pragma unroll
    for (int s = 0; s < 4; ++s)
        #pragma unroll
        for (int t = 0; t < 2; ++t)
            #pragma unroll
            for (int v = 0; v < 8; ++v) {
                int m = m0 + wm * 64 + s * 16 + v + 8 * g;
                int n = n0 + wn * 32 + t * 16 + lr;
                size_t idx = (size_t)m * N + n;
                float a = acc[s][t][v];
                if (MODE == 0) outF[idx] = a;
                else if (MODE == 1) outF[idx] = a + res[idx];
                else { // silu(a) * b
                    float b = acc2[s][t][v];
                    float sg = a / (1.0f + __expf(-a));
                    outBf[idx] = f2bf(sg * b);
                }
            }
}

// ---------------- RoPE + layout: qkv f32 -> q/k bf16 [bh,s,hd], v^T bf16 [bkh,hd,s]
// NOTE: softmax scale 1/sqrt(HD) = 0.125 is folded into Q here.
__global__ void rope_split(const float* __restrict__ qkv, const float* __restrict__ freqs,
                           u16* __restrict__ qo, u16* __restrict__ ko, u16* __restrict__ vT) {
    int tok = blockIdx.x;               // b*S + s
    int b = tok / S_, s = tok % S_;
    const float* row = qkv + (size_t)tok * NQKV_;
    const float* fr = freqs + (size_t)s * HD_;   // [32][2] cos,sin
    for (int idx = threadIdx.x; idx < H_ * (HD_ / 2); idx += 256) {
        int h = idx >> 5, p = idx & 31;
        float c = fr[2 * p], sn = fr[2 * p + 1];
        float x0 = row[h * HD_ + 2 * p], x1 = row[h * HD_ + 2 * p + 1];
        size_t o = ((size_t)(b * H_ + h) * S_ + s) * HD_ + 2 * p;
        qo[o] = f2bf((x0 * c - x1 * sn) * 0.125f);
        qo[o + 1] = f2bf((x1 * c + x0 * sn) * 0.125f);
    }
    for (int idx = threadIdx.x; idx < KH_ * (HD_ / 2); idx += 256) {
        int kh = idx >> 5, p = idx & 31;
        float c = fr[2 * p], sn = fr[2 * p + 1];
        float x0 = row[E_ + kh * HD_ + 2 * p], x1 = row[E_ + kh * HD_ + 2 * p + 1];
        size_t o = ((size_t)(b * KH_ + kh) * S_ + s) * HD_ + 2 * p;
        ko[o] = f2bf(x0 * c - x1 * sn);
        ko[o + 1] = f2bf(x1 * c + x0 * sn);
    }
    for (int idx = threadIdx.x; idx < KH_ * HD_; idx += 256) {
        int kh = idx >> 6, d = idx & 63;
        vT[((size_t)(b * KH_ + kh) * HD_ + d) * S_ + s] = f2bf(row[E_ + KV_ + kh * HD_ + d]);
    }
}

// ---------------- Flash attention (causal, GQA rep=4), WMMA bf16 --------------
template <bool MASKED>
static __device__ __forceinline__ void attn_block(
    int kb, int q0, int g, int lr,
    const u16* __restrict__ krow, const u16* __restrict__ vt,
    const v16bf (&qf)[2], v8f (&oacc)[4], float (&rmax)[8], float (&rsum)[8],
    u16 (*Pw)[40]) {
    v8f sc[2] = {};
    #pragma unroll
    for (int jj = 0; jj < 2; ++jj) {
        const u16* kr = krow + (size_t)(kb + jj * 16 + lr) * HD_;
        #pragma unroll
        for (int kc = 0; kc < 2; ++kc) {
            v16bf kf = mk16(ld8(kr + kc * 32 + 16 * g), ld8(kr + kc * 32 + 16 * g + 8));
            sc[jj] = wmma_bf(qf[kc], kf, sc[jj]);
        }
    }
    #pragma unroll
    for (int v = 0; v < 8; ++v) {
        float s0 = sc[0][v], s1 = sc[1][v];     // scale pre-folded into Q
        if (MASKED) {
            int qrow = q0 + v + 8 * g;
            if (kb + lr > qrow) s0 = -1e30f;
            if (kb + 16 + lr > qrow) s1 = -1e30f;
        }
        float mv = red16_max(fmaxf(s0, s1));
        float nm = fmaxf(rmax[v], mv);
        float corr = __expf(rmax[v] - nm);
        rmax[v] = nm;
        float p0 = __expf(s0 - nm), p1 = __expf(s1 - nm);
        float ps = red16_sum(p0 + p1);
        rsum[v] = rsum[v] * corr + ps;
        #pragma unroll
        for (int t = 0; t < 4; ++t) oacc[t][v] *= corr;
        Pw[v + 8 * g][lr] = f2bf(p0);
        Pw[v + 8 * g][16 + lr] = f2bf(p1);
    }
    v16bf pf = mk16(ld8(&Pw[lr][16 * g]), ld8(&Pw[lr][16 * g + 8]));
    #pragma unroll
    for (int t = 0; t < 4; ++t) {
        const u16* vr = vt + (size_t)(t * 16 + lr) * S_ + kb + 16 * g;
        v16bf vf = mk16(ld8(vr), ld8(vr + 8));
        oacc[t] = wmma_bf(pf, vf, oacc[t]);
    }
}

__global__ __launch_bounds__(256) void attn_flash(
    const u16* __restrict__ q, const u16* __restrict__ k, const u16* __restrict__ vT,
    u16* __restrict__ o) {
    __shared__ __align__(16) u16 P[8][16][40];   // per-wave P tile, padded
    const int bh = blockIdx.y;
    const int b = bh / H_, h = bh % H_;
    const int bk = b * KH_ + (h / 4);
    const int wave = threadIdx.x >> 5, lane = threadIdx.x & 31;
    const int g = lane >> 4, lr = lane & 15;
    const int q0 = blockIdx.x * 128 + wave * 16;
    const u16* krow = k + (size_t)bk * S_ * HD_;
    const u16* vt = vT + (size_t)bk * HD_ * S_;

    v16bf qf[2];
    {
        const u16* qb = q + ((size_t)bh * S_ + q0 + lr) * HD_;
        #pragma unroll
        for (int kc = 0; kc < 2; ++kc)
            qf[kc] = mk16(ld8(qb + kc * 32 + 16 * g), ld8(qb + kc * 32 + 16 * g + 8));
    }

    v8f oacc[4] = {};
    float rmax[8], rsum[8];
    #pragma unroll
    for (int v = 0; v < 8; ++v) { rmax[v] = -1e30f; rsum[v] = 0.f; }

    const int kend = q0 + 16;                               // keys < q0+16 (causal)
    const int kfull = (q0 >= 31) ? ((((q0 - 31) >> 5) << 5) + 32) : 0;  // fully unmasked prefix
    for (int kb = 0; kb < kfull; kb += 32)
        attn_block<false>(kb, q0, g, lr, krow, vt, qf, oacc, rmax, rsum, P[wave]);
    for (int kb = kfull; kb < kend; kb += 32)
        attn_block<true>(kb, q0, g, lr, krow, vt, qf, oacc, rmax, rsum, P[wave]);

    #pragma unroll
    for (int t = 0; t < 4; ++t)
        #pragma unroll
        for (int v = 0; v < 8; ++v) {
            int qrow = q0 + v + 8 * g;
            int col = h * HD_ + t * 16 + lr;
            o[((size_t)b * S_ + qrow) * E_ + col] = f2bf(oacc[t][v] / rsum[v]);
        }
}

// =====================================================================
extern "C" void kernel_launch(void* const* d_in, const int* in_sizes, int n_in,
                              void* d_out, int out_size, void* d_ws, size_t ws_size,
                              hipStream_t stream) {
    const float* x      = (const float*)d_in[0];
    const float* freqs  = (const float*)d_in[2];
    const float* w_qkv  = (const float*)d_in[3];
    const float* w_o    = (const float*)d_in[4];
    const float* w1     = (const float*)d_in[5];
    const float* w2     = (const float*)d_in[6];
    const float* w3     = (const float*)d_in[7];
    const float* anw    = (const float*)d_in[8];
    const float* fnw    = (const float*)d_in[9];
    float* out = (float*)d_out;

    char* ws = (char*)d_ws;
    size_t off = 0;
    auto alloc = [&](size_t bytes) { void* p = ws + off; off = (off + bytes + 255) & ~(size_t)255; return p; };

    u16* wqkv_bf = (u16*)alloc((size_t)NQKV_ * E_ * 2);
    u16* wo_bf   = (u16*)alloc((size_t)E_ * E_ * 2);
    u16* w1_bf   = (u16*)alloc((size_t)F_ * E_ * 2);
    u16* w3_bf   = (u16*)alloc((size_t)F_ * E_ * 2);
    u16* w2_bf   = (u16*)alloc((size_t)E_ * F_ * 2);
    u16* h_bf    = (u16*)alloc((size_t)M_ * E_ * 2);      // reused as g_bf
    float* qkv_f = (float*)alloc((size_t)M_ * NQKV_ * 4); // reused as h2
    u16* q_bf    = (u16*)alloc((size_t)B_ * H_ * S_ * HD_ * 2);
    u16* k_bf    = (u16*)alloc((size_t)B_ * KH_ * S_ * HD_ * 2);
    u16* vT_bf   = (u16*)alloc((size_t)B_ * KH_ * HD_ * S_ * 2);
    u16* ao_bf   = (u16*)alloc((size_t)M_ * E_ * 2);
    u16* gate_bf = (u16*)alloc((size_t)M_ * F_ * 2);
    float* h2 = qkv_f;   // overlay: qkv dead after rope_split
    u16* g_bf = h_bf;    // overlay: h dead after qkv GEMM

    cvt_f32_bf16<<<2048, 256, 0, stream>>>(w_qkv, wqkv_bf, NQKV_ * E_ / 4);
    cvt_f32_bf16<<<2048, 256, 0, stream>>>(w_o,   wo_bf,   E_ * E_ / 4);
    cvt_f32_bf16<<<4096, 256, 0, stream>>>(w1,    w1_bf,   F_ * E_ / 4);
    cvt_f32_bf16<<<4096, 256, 0, stream>>>(w3,    w3_bf,   F_ * E_ / 4);
    cvt_f32_bf16<<<4096, 256, 0, stream>>>(w2,    w2_bf,   E_ * F_ / 4);

    rmsnorm_bf16<<<M_, 256, 0, stream>>>(x, anw, h_bf);

    gemm_bf16<0, false><<<dim3(NQKV_ / 128, M_ / 128), 256, 0, stream>>>(
        h_bf, wqkv_bf, nullptr, nullptr, qkv_f, nullptr, M_, NQKV_, E_);

    rope_split<<<M_, 256, 0, stream>>>(qkv_f, freqs, q_bf, k_bf, vT_bf);

    attn_flash<<<dim3(S_ / 128, B_ * H_), 256, 0, stream>>>(q_bf, k_bf, vT_bf, ao_bf);

    gemm_bf16<1, false><<<dim3(E_ / 128, M_ / 128), 256, 0, stream>>>(
        ao_bf, wo_bf, nullptr, x, h2, nullptr, M_, E_, E_);

    rmsnorm_bf16<<<M_, 256, 0, stream>>>(h2, fnw, g_bf);

    gemm_bf16<2, true><<<dim3(F_ / 128, M_ / 128), 256, 0, stream>>>(
        g_bf, w1_bf, w3_bf, nullptr, nullptr, gate_bf, M_, F_, E_);

    gemm_bf16<1, false><<<dim3(E_ / 128, M_ / 128), 256, 0, stream>>>(
        gate_bf, w2_bf, nullptr, h2, out, nullptr, M_, E_, F_);

    (void)in_sizes; (void)n_in; (void)out_size; (void)ws_size;
}
